// GNNEncoder_16716012716418
// MI455X (gfx1250) — compile-verified, matrix operand order checked by reading, not run on hardware
//
#include <hip/hip_runtime.h>
#include <hip/hip_bf16.h>
#include <stdint.h>

// ---------------------------------------------------------------------------
// GINE-style GNN encoder for MI455X (gfx1250, wave32, WMMA + TDM both ways).
//   x = nf @ Win + b_in                      (x stored bf16: halves gather BW)
//   repeat 3x:  agg = x + scatter_add(relu(x[src] + attr*We + be), dst)  (f32)
//               x   = relu( relu(agg@W1+b1) @ W2 + b2 )
// GEMMs: v_wmma_f32_16x16x32_bf16. Input tiles DMA'd global->LDS and result
// tiles DMA'd LDS->global by the Tensor Data Mover (OOB rows handled by D#).
// ---------------------------------------------------------------------------

typedef __attribute__((ext_vector_type(16))) __bf16        v16bf;
typedef __attribute__((ext_vector_type(8)))  float         v8f;
typedef __attribute__((ext_vector_type(4)))  unsigned int  u32x4;
typedef __attribute__((ext_vector_type(2)))  unsigned int  u32x2;
typedef __attribute__((ext_vector_type(4)))  float         f32x4;
typedef __attribute__((ext_vector_type(8)))  int           i32x8;
typedef __attribute__((ext_vector_type(4)))  int           i32x4;

__device__ __forceinline__ unsigned short f32_to_bf16(float f) {
  unsigned int u = __float_as_uint(f);
  u += 0x7fffu + ((u >> 16) & 1u);       // round-to-nearest-even
  return (unsigned short)(u >> 16);
}
__device__ __forceinline__ float bf16_to_f32(unsigned int h) {
  return __uint_as_float(h << 16);
}

// ---------------------------------------------------------------------------
// TDM 2D tile descriptor (CDNA5 ISA §8.3-8.5): 16-row x cols tile, row-major,
// stride == cols, element size by dsizeCode (1 -> 2B, 2 -> 4B).
// tensor_dim1 = rows remaining: OOB rows load as zero / store dropped.
// Issue from ONE wave only (EXEC ignored by TDM). 6-arg builtin flavor.
// ---------------------------------------------------------------------------
__device__ __forceinline__ void tdm_desc(u32x4& g0, i32x8& g1,
                                         const void* gaddr, unsigned ldsOff,
                                         unsigned rows, unsigned cols,
                                         unsigned dsizeCode) {
  unsigned long long ga = (unsigned long long)(size_t)gaddr;
  g0[0] = 1u;                                          // count=1, user D#
  g0[1] = ldsOff;                                      // lds_addr
  g0[2] = (unsigned)ga;                                // global_addr[31:0]
  g0[3] = (unsigned)((ga >> 32) & 0x01ffffffu)         // global_addr[56:32]
        | (2u << 30);                                  // type=2 ("image")
  g1[0] = (int)(dsizeCode << 16);                      // data_size
  g1[1] = (int)(cols << 16);                           // tensor_dim0[15:0]
  g1[2] = (int)((rows & 0xffffu) << 16);               // dim0 hi=0 | tensor_dim1 lo
  g1[3] = (int)((rows >> 16) | (cols << 16));          // tensor_dim1 hi | tile_dim0
  g1[4] = 16;                                          // tile_dim1 = 16 rows
  g1[5] = (int)cols;                                   // tensor_dim0_stride lo
  g1[6] = 0;
  g1[7] = 0;
}
__device__ __forceinline__ void tdm_load_tile(const void* gsrc, unsigned ldsOff,
                                              unsigned rows, unsigned cols,
                                              unsigned dsizeCode) {
  u32x4 g0; i32x8 g1;
  tdm_desc(g0, g1, gsrc, ldsOff, rows, cols, dsizeCode);
  i32x4 z4 = {0, 0, 0, 0};
  i32x8 z8 = {0, 0, 0, 0, 0, 0, 0, 0};
  __builtin_amdgcn_tensor_load_to_lds(g0, g1, z4, z4, z8, 0);
}
__device__ __forceinline__ void tdm_store_tile(const void* gdst, unsigned ldsOff,
                                               unsigned rows, unsigned cols,
                                               unsigned dsizeCode) {
  u32x4 g0; i32x8 g1;
  tdm_desc(g0, g1, gdst, ldsOff, rows, cols, dsizeCode);
  i32x4 z4 = {0, 0, 0, 0};
  i32x8 z8 = {0, 0, 0, 0, 0, 0, 0, 0};
  __builtin_amdgcn_tensor_store_from_lds(g0, g1, z4, z4, z8, 0);
}

// ---------------------------------------------------------------------------
// Pack a KxN f32 weight matrix into per-lane WMMA B fragments (bf16 pairs).
// ISA 16-bit layout: lane = (n%16) + 16*half, dword v holds K-pair
//   k = kt*32 + (v%4)*2 + half*8 + (v/4)*16.
// Bp[((nTile*KT + kt)*32 + lane)*8 + v] -> 32B contiguous per lane.
// ---------------------------------------------------------------------------
__global__ void pack_b_kernel(const float* __restrict__ W,
                              unsigned int* __restrict__ Bp, int K, int N) {
  int wave = (blockIdx.x * blockDim.x + threadIdx.x) >> 5;
  int lane = threadIdx.x & 31;
  int KT = K >> 5, NT = N >> 4;
  if (wave >= KT * NT) return;
  int nTile = wave / KT, kt = wave % KT;
  int n  = nTile * 16 + (lane & 15);
  int hf = lane >> 4;
  unsigned int* dst = Bp + ((size_t)(nTile * KT + kt) * 32 + lane) * 8;
#pragma unroll
  for (int v = 0; v < 8; ++v) {
    int k = kt * 32 + (v & 3) * 2 + hf * 8 + (v >> 2) * 16;
    unsigned int lo = f32_to_bf16(W[(size_t)k * N + n]);
    unsigned int hi = f32_to_bf16(W[(size_t)(k + 1) * N + n]);
    dst[v] = lo | (hi << 16);
  }
}

// A fragment from LDS row-major bf16 tile: lane row = lane%16, half = lane/16.
__device__ __forceinline__ v16bf load_a_frag(const unsigned short* sRow,
                                             int kt, int hf) {
  v16bf a;
  const char* base = (const char*)sRow;
  *((u32x4*)&a)     = *(const u32x4*)(base + kt * 64 + hf * 16);
  *((u32x4*)&a + 1) = *(const u32x4*)(base + kt * 64 + 32 + hf * 16);
  return a;
}
__device__ __forceinline__ v16bf load_b_frag(const unsigned int* __restrict__ Bp,
                                             int nTile, int KT, int kt, int lane) {
  v16bf b;
  const u32x4* p = (const u32x4*)(Bp + ((size_t)(nTile * KT + kt) * 32 + lane) * 8);
  *((u32x4*)&b)     = p[0];
  *((u32x4*)&b + 1) = p[1];
  return b;
}

// ---------------------------------------------------------------------------
// Input projection: xbf16[M,128] = nf[M,64] @ WinPacked + b_in.  K=64 fixed.
// Block: 16 rows; 256 threads = 8 waves; wave w -> N-tile w.
// TDM-staged A in, TDM-stored bf16 result out (sStage region reused).
// ---------------------------------------------------------------------------
__global__ void proj_kernel(const float* __restrict__ A,
                            const unsigned int* __restrict__ Bp,
                            const float* __restrict__ bias,
                            unsigned short* __restrict__ out, int M) {
  __shared__ __align__(16) float          sStage[16 * 64];   // 4KB; reused as bf16 out
  __shared__ __align__(16) unsigned short sA[16 * 64];       // 2KB
  int row0 = blockIdx.x * 16;
  int tid  = threadIdx.x;
  unsigned stageOff = (unsigned)(size_t)sStage;

  if (tid < 32) {                                     // wave 0 issues the DMA
    tdm_load_tile(A + (size_t)row0 * 64, stageOff, (unsigned)(M - row0), 64u, 2u);
    __builtin_amdgcn_s_wait_tensorcnt(0);
  }
  __syncthreads();
  for (int i = tid; i < 16 * 64; i += 256) sA[i] = f32_to_bf16(sStage[i]);
  __syncthreads();                                    // sStage dead after this

  int wave = tid >> 5, lane = tid & 31;
  int mA = lane & 15, hf = lane >> 4;
  const unsigned short* sRow = sA + (size_t)mA * 64;

  v8f acc = {};
#pragma unroll
  for (int kt = 0; kt < 2; ++kt) {
    if (kt < 1)
      __builtin_prefetch((const void*)(Bp + ((size_t)(wave * 2 + kt + 1) * 32 + lane) * 8), 0, 1);
    v16bf a = load_a_frag(sRow, kt, hf);
    v16bf b = load_b_frag(Bp, wave, 2, kt, lane);
    acc = __builtin_amdgcn_wmma_f32_16x16x32_bf16(false, a, false, b,
                                                  (short)0, acc, false, false);
  }
  int n = wave * 16 + mA;
  float bn = bias[n];
  unsigned short* sOut = (unsigned short*)sStage;     // 16x128 bf16 = 4KB
#pragma unroll
  for (int v = 0; v < 8; ++v) {
    int r = v + hf * 8;                               // C layout: m = v + 8*half
    sOut[r * 128 + n] = f32_to_bf16(acc[v] + bn);
  }
  __syncthreads();
  if (tid < 32)                                       // DMA result tile out
    tdm_store_tile(out + (size_t)row0 * 128, stageOff, (unsigned)(M - row0), 128u, 1u);
  // S_ENDPGM implicitly waits for TENSORcnt
}

// agg(f32) = x(bf16)   (init before scatter-add); 4 elements/thread
__global__ void copy_bf16_to_f32_kernel(const unsigned short* __restrict__ src,
                                        float* __restrict__ dst, size_t n4) {
  size_t i = (size_t)blockIdx.x * blockDim.x + threadIdx.x;
  if (i >= n4) return;
  u32x2 s = ((const u32x2*)src)[i];
  f32x4 d;
  d.x = bf16_to_f32(s[0] & 0xffffu);
  d.y = bf16_to_f32(s[0] >> 16);
  d.z = bf16_to_f32(s[1] & 0xffffu);
  d.w = bf16_to_f32(s[1] >> 16);
  ((f32x4*)dst)[i] = d;
}

// ---------------------------------------------------------------------------
// One wave per edge: agg[dst] += relu(x[src] + attr*We + be)  (128 cols).
// x is bf16 (8B/lane gather); agg accumulated with f32 atomics.
// ---------------------------------------------------------------------------
__global__ void edge_scatter_kernel(const unsigned short* __restrict__ x,
                                    const long long* __restrict__ ei,
                                    const float* __restrict__ attr,
                                    const float* __restrict__ We,
                                    const float* __restrict__ be,
                                    float* __restrict__ agg, int E) {
  int e    = (int)(((size_t)blockIdx.x * blockDim.x + threadIdx.x) >> 5);
  int lane = threadIdx.x & 31;
  if (e >= E) return;
  long long s = ei[e];
  long long d = ei[(size_t)E + e];
  float a = attr[e];
  int c = lane * 4;
  u32x2 xb = *(const u32x2*)(x + (size_t)s * 128 + c);
  f32x4 wv = *(const f32x4*)(We + c);
  f32x4 bv = *(const f32x4*)(be + c);
  float m0 = fmaxf(bf16_to_f32(xb[0] & 0xffffu) + a * wv.x + bv.x, 0.0f);
  float m1 = fmaxf(bf16_to_f32(xb[0] >> 16)     + a * wv.y + bv.y, 0.0f);
  float m2 = fmaxf(bf16_to_f32(xb[1] & 0xffffu) + a * wv.z + bv.z, 0.0f);
  float m3 = fmaxf(bf16_to_f32(xb[1] >> 16)     + a * wv.w + bv.w, 0.0f);
  float* p = agg + (size_t)d * 128 + c;
  atomicAdd(p + 0, m0);
  atomicAdd(p + 1, m1);
  atomicAdd(p + 2, m2);
  atomicAdd(p + 3, m3);
}

// ---------------------------------------------------------------------------
// Fused layer MLP: out = relu( relu(Hin@W1+b1) @ W2 + b2 ),  H = 128.
// Hin tile DMA'd in by TDM; intermediate in LDS (bf16); result staged in the
// (dead) sStage region and DMA'd out by TDM (bf16 layers 0..1, f32 last).
// ---------------------------------------------------------------------------
__global__ void mlp_kernel(const float* __restrict__ Hin,
                           const unsigned int* __restrict__ B1p,
                           const float* __restrict__ b1,
                           const unsigned int* __restrict__ B2p,
                           const float* __restrict__ b2,
                           void* __restrict__ outv, int out_is_bf16, int M) {
  __shared__ __align__(16) float          sStage[16 * 128];  // 8KB; reused for out
  __shared__ __align__(16) unsigned short sA[16 * 128];      // 4KB
  __shared__ __align__(16) unsigned short sT[16 * 128];      // 4KB
  int row0 = blockIdx.x * 16;
  int tid  = threadIdx.x;
  unsigned stageOff = (unsigned)(size_t)sStage;

  if (tid < 32) {                                     // wave 0 issues the DMA
    tdm_load_tile(Hin + (size_t)row0 * 128, stageOff, (unsigned)(M - row0), 128u, 2u);
    __builtin_amdgcn_s_wait_tensorcnt(0);
  }
  __syncthreads();
  for (int i = tid; i < 2048; i += 256) sA[i] = f32_to_bf16(sStage[i]);
  __syncthreads();                                    // sStage dead after this

  int wave = tid >> 5, lane = tid & 31;
  int mA = lane & 15, hf = lane >> 4;
  int n  = wave * 16 + mA;

  // GEMM 1: t = relu(Hin @ W1 + b1)
  v8f acc = {};
#pragma unroll
  for (int kt = 0; kt < 4; ++kt) {
    if (kt < 3)
      __builtin_prefetch((const void*)(B1p + ((size_t)(wave * 4 + kt + 1) * 32 + lane) * 8), 0, 1);
    v16bf a = load_a_frag(sA + (size_t)mA * 128, kt, hf);
    v16bf b = load_b_frag(B1p, wave, 4, kt, lane);
    acc = __builtin_amdgcn_wmma_f32_16x16x32_bf16(false, a, false, b,
                                                  (short)0, acc, false, false);
  }
  float bn1 = b1[n];
#pragma unroll
  for (int v = 0; v < 8; ++v) {
    float t = fmaxf(acc[v] + bn1, 0.0f);
    int r = v + hf * 8;                     // C layout: m = v + 8*half
    sT[(size_t)r * 128 + n] = f32_to_bf16(t);
  }
  __syncthreads();

  // GEMM 2: out = relu(t @ W2 + b2)
  v8f acc2 = {};
#pragma unroll
  for (int kt = 0; kt < 4; ++kt) {
    if (kt < 3)
      __builtin_prefetch((const void*)(B2p + ((size_t)(wave * 4 + kt + 1) * 32 + lane) * 8), 0, 1);
    v16bf a = load_a_frag(sT + (size_t)mA * 128, kt, hf);
    v16bf b = load_b_frag(B2p, wave, 4, kt, lane);
    acc2 = __builtin_amdgcn_wmma_f32_16x16x32_bf16(false, a, false, b,
                                                   (short)0, acc2, false, false);
  }
  float bn2 = b2[n];
  if (out_is_bf16) {
    unsigned short* sOut = (unsigned short*)sStage;    // 16x128 bf16 = 4KB
#pragma unroll
    for (int v = 0; v < 8; ++v) {
      float t = fmaxf(acc2[v] + bn2, 0.0f);
      int r = v + hf * 8;
      sOut[r * 128 + n] = f32_to_bf16(t);
    }
  } else {
    float* sOut = sStage;                              // 16x128 f32 = 8KB
#pragma unroll
    for (int v = 0; v < 8; ++v) {
      float t = fmaxf(acc2[v] + bn2, 0.0f);
      int r = v + hf * 8;
      sOut[r * 128 + n] = t;
    }
  }
  __syncthreads();
  if (tid < 32) {                                      // DMA result tile out
    if (out_is_bf16)
      tdm_store_tile((unsigned short*)outv + (size_t)row0 * 128, stageOff,
                     (unsigned)(M - row0), 128u, 1u);
    else
      tdm_store_tile((float*)outv + (size_t)row0 * 128, stageOff,
                     (unsigned)(M - row0), 128u, 2u);
  }
  // S_ENDPGM implicitly waits for TENSORcnt
}

// ---------------------------------------------------------------------------
extern "C" void kernel_launch(void* const* d_in, const int* in_sizes, int n_in,
                              void* d_out, int out_size, void* d_ws, size_t ws_size,
                              hipStream_t stream) {
  const float*     node = (const float*)d_in[0];
  const long long* ei   = (const long long*)d_in[1];   // int64 edge_index [2,E]
  const float*     attr = (const float*)d_in[2];
  const float*     Win  = (const float*)d_in[3];
  const float*     b_in = (const float*)d_in[4];
  const float*     We   = (const float*)d_in[5];
  const float*     be   = (const float*)d_in[6];
  const float*     W1   = (const float*)d_in[7];
  const float*     b1   = (const float*)d_in[8];
  const float*     W2   = (const float*)d_in[9];
  const float*     b2   = (const float*)d_in[10];

  const int H = 128, DIN = 64;
  int N = in_sizes[0] / DIN;
  int E = in_sizes[2];

  // Workspace carve-up
  char* ws = (char*)d_ws;
  unsigned short* xbuf = (unsigned short*)ws;  ws += (size_t)N * H * sizeof(unsigned short);
  float* aggbuf = (float*)ws;                  ws += (size_t)N * H * sizeof(float);
  unsigned int* WinP = (unsigned int*)ws;  ws += (size_t)8 * 2 * 32 * 8 * 4;     // 16 KB
  unsigned int* W1P  = (unsigned int*)ws;  ws += (size_t)3 * 8 * 4 * 32 * 8 * 4; // 96 KB
  unsigned int* W2P  = (unsigned int*)ws;  ws += (size_t)3 * 8 * 4 * 32 * 8 * 4; // 96 KB

  // Pack weights into WMMA B-fragment layout (bf16)
  pack_b_kernel<<<2, 256, 0, stream>>>(Win, WinP, DIN, H);           // 16 waves
  for (int l = 0; l < 3; ++l) {
    pack_b_kernel<<<4, 256, 0, stream>>>(W1 + (size_t)l * H * H, W1P + (size_t)l * 8192, H, H);
    pack_b_kernel<<<4, 256, 0, stream>>>(W2 + (size_t)l * H * H, W2P + (size_t)l * 8192, H, H);
  }

  int mTiles = (N + 15) / 16;                                        // 3125
  proj_kernel<<<mTiles, 256, 0, stream>>>(node, WinP, b_in, xbuf, N);

  size_t n4 = (size_t)N * H / 4;
  int copyBlocks = (int)((n4 + 255) / 256);
  int edgeBlocks = (E + 7) / 8;                                      // 8 edges/block

  for (int l = 0; l < 3; ++l) {
    copy_bf16_to_f32_kernel<<<copyBlocks, 256, 0, stream>>>(xbuf, aggbuf, n4);
    edge_scatter_kernel<<<edgeBlocks, 256, 0, stream>>>(xbuf, ei, attr, We, be, aggbuf, E);
    int last = (l == 2);
    void* outp = last ? d_out : (void*)xbuf;
    mlp_kernel<<<mTiles, 256, 0, stream>>>(aggbuf,
                                           W1P + (size_t)l * 8192, b1 + (size_t)l * H,
                                           W2P + (size_t)l * 8192, b2 + (size_t)l * H,
                                           outp, !last, N);
  }
}
